// W_CA_Module_86217173500261
// MI455X (gfx1250) — matrix operand order repaired and checked by reading, not compiled, not run on hardware
//
#include <hip/hip_runtime.h>
#include <hip/hip_bf16.h>

// ---------------------------------------------------------------------------
// CDNA5 (gfx1250) implementation of the banded cross-attention module.
// All matmuls go through v_wmma_f32_16x16x32_bf16 (wave32 WMMA, f32 accum).
// Attention K-tiles staged to LDS via global_load_async_to_lds_b128 (ASYNCcnt).
// ---------------------------------------------------------------------------

typedef __attribute__((ext_vector_type(16))) __bf16 v16bf;
typedef __attribute__((ext_vector_type(8)))  float  v8f;

union FragAB {
  v16bf bf;
  uint4 u4[2];
  unsigned int u[8];
};

static __device__ inline v8f vzero() {
  v8f z = {0.f,0.f,0.f,0.f,0.f,0.f,0.f,0.f};
  return z;
}

static __device__ inline v8f wmma_bf16(const FragAB& a, const FragAB& b, v8f c) {
  return __builtin_amdgcn_wmma_f32_16x16x32_bf16(
      /*neg_a=*/false, a.bf, /*neg_b=*/false, b.bf,
      /*c_mod=*/(short)0, c, /*reuse_a=*/false, /*reuse_b=*/false);
}

static __device__ inline unsigned short f2bf(float f) {
  unsigned int u = __float_as_uint(f);
  unsigned int r = u + 0x7FFFu + ((u >> 16) & 1u);   // round-to-nearest-even
  return (unsigned short)(r >> 16);
}
static __device__ inline float bf2f(unsigned short h) {
  return __uint_as_float(((unsigned int)h) << 16);
}

// 64-byte async copy global -> LDS, no VGPR round-trip (tracked by ASYNCcnt).
static __device__ inline void async_copy_64B(unsigned lds_off, const void* gptr) {
  unsigned long long ga = (unsigned long long)gptr;
  asm volatile(
      "global_load_async_to_lds_b128 %0, %1, off\n\t"
      "global_load_async_to_lds_b128 %0, %1, off offset:16\n\t"
      "global_load_async_to_lds_b128 %0, %1, off offset:32\n\t"
      "global_load_async_to_lds_b128 %0, %1, off offset:48"
      :: "v"(lds_off), "v"(ga) : "memory");
}
static __device__ inline void wait_async0() {
  asm volatile("s_wait_asynccnt 0x0" ::: "memory");
}

#define CC   256
#define HH   64
#define WW   512
#define BB   2
#define MM   65536        // B*H*W
#define NN   128          // B*H
#define BAND 133          // int(512 * 0.26)

// ---------------------------------------------------------------------------
// 1) f32 (B,C,H,W) -> bf16 (M, C) transpose (channels contiguous)
// ---------------------------------------------------------------------------
__global__ void transpose_to_bf16(const float* __restrict__ x,
                                  unsigned short* __restrict__ xt) {
  __shared__ float tile[32][33];
  const int bh = blockIdx.z;                 // 0..127
  const int b = bh >> 6, h = bh & 63;
  const int c0 = blockIdx.y * 32;
  const int w0 = blockIdx.x * 32;
  const int tx = threadIdx.x, ty = threadIdx.y;
  #pragma unroll
  for (int i = 0; i < 4; ++i) {
    int c = c0 + ty + i * 8;
    tile[ty + i * 8][tx] = x[(((size_t)b * CC + c) * HH + h) * WW + w0 + tx];
  }
  __syncthreads();
  #pragma unroll
  for (int i = 0; i < 4; ++i) {
    int wl = w0 + ty + i * 8;
    xt[((size_t)bh * WW + wl) * CC + c0 + tx] = f2bf(tile[tx][ty + i * 8]);
  }
}

// ---------------------------------------------------------------------------
// 2) weight conversion f32 -> bf16 (4 matrices, 256x256 each)
// ---------------------------------------------------------------------------
__global__ void convert_weights(const float* __restrict__ q, const float* __restrict__ k,
                                const float* __restrict__ v, const float* __restrict__ w,
                                unsigned short* __restrict__ oq, unsigned short* __restrict__ ok,
                                unsigned short* __restrict__ ov, unsigned short* __restrict__ ow) {
  int i = blockIdx.x * blockDim.x + threadIdx.x;   // 65536 threads
  oq[i] = f2bf(q[i]);
  ok[i] = f2bf(k[i]);
  ov[i] = f2bf(v[i]);
  ow[i] = f2bf(w[i]);
}

// ---------------------------------------------------------------------------
// 3) fused Q/K/V 1x1-conv GEMM: out[m,o] = sum_c xt[m,c]*W[o,c] + b[o]
// ---------------------------------------------------------------------------
__global__ __launch_bounds__(256) void qkv_gemm(
    const unsigned short* __restrict__ xt,
    const unsigned short* __restrict__ wq, const float* __restrict__ bq, unsigned short* __restrict__ oq,
    const unsigned short* __restrict__ wk, const float* __restrict__ bk, unsigned short* __restrict__ ok,
    const unsigned short* __restrict__ wv, const float* __restrict__ bv, unsigned short* __restrict__ ov) {
  const int wave = blockIdx.x * 8 + (threadIdx.x >> 5);
  const int lane = threadIdx.x & 31;
  const int m0 = wave * 16;
  const int nk = lane & 15;
  const int hi = (lane < 16) ? 0 : 1;

  FragAB a[8];
  {
    const uint4* rp = (const uint4*)(xt + ((size_t)m0 + nk) * CC);
    #pragma unroll
    for (int kc = 0; kc < 8; ++kc) {
      a[kc].u4[0] = rp[kc * 4 + hi * 2];
      a[kc].u4[1] = rp[kc * 4 + hi * 2 + 1];
    }
  }
  const unsigned short* wsets[3] = {wq, wk, wv};
  const float*          bsets[3] = {bq, bk, bv};
  unsigned short*       osets[3] = {oq, ok, ov};

  #pragma unroll
  for (int s = 0; s < 3; ++s) {
    const unsigned short* wmat = wsets[s];
    const float* bias = bsets[s];
    unsigned short* out = osets[s];
    for (int og = 0; og < 16; ++og) {
      const int ocol = og * 16 + nk;
      v8f acc = vzero();
      const uint4* wr = (const uint4*)(wmat + (size_t)ocol * CC);
      #pragma unroll
      for (int kc = 0; kc < 8; ++kc) {
        FragAB bfr;
        bfr.u4[0] = wr[kc * 4 + hi * 2];
        bfr.u4[1] = wr[kc * 4 + hi * 2 + 1];
        acc = wmma_bf16(a[kc], bfr, acc);
      }
      const float bs = bias[ocol];
      const int rbase = m0 + hi * 8;
      #pragma unroll
      for (int r = 0; r < 8; ++r)
        out[(size_t)(rbase + r) * CC + ocol] = f2bf(acc[r] + bs);
    }
  }
}

// ---------------------------------------------------------------------------
// 4) single-weight GEMM (final W projection)
// ---------------------------------------------------------------------------
__global__ __launch_bounds__(256) void gemm_one(
    const unsigned short* __restrict__ xt, const unsigned short* __restrict__ wmat,
    const float* __restrict__ bias, unsigned short* __restrict__ out) {
  const int wave = blockIdx.x * 8 + (threadIdx.x >> 5);
  const int lane = threadIdx.x & 31;
  const int m0 = wave * 16;
  const int nk = lane & 15;
  const int hi = (lane < 16) ? 0 : 1;

  FragAB a[8];
  {
    const uint4* rp = (const uint4*)(xt + ((size_t)m0 + nk) * CC);
    #pragma unroll
    for (int kc = 0; kc < 8; ++kc) {
      a[kc].u4[0] = rp[kc * 4 + hi * 2];
      a[kc].u4[1] = rp[kc * 4 + hi * 2 + 1];
    }
  }
  for (int og = 0; og < 16; ++og) {
    const int ocol = og * 16 + nk;
    v8f acc = vzero();
    const uint4* wr = (const uint4*)(wmat + (size_t)ocol * CC);
    #pragma unroll
    for (int kc = 0; kc < 8; ++kc) {
      FragAB bfr;
      bfr.u4[0] = wr[kc * 4 + hi * 2];
      bfr.u4[1] = wr[kc * 4 + hi * 2 + 1];
      acc = wmma_bf16(a[kc], bfr, acc);
    }
    const float bs = bias[ocol];
    const int rbase = m0 + hi * 8;
    #pragma unroll
    for (int r = 0; r < 8; ++r)
      out[(size_t)(rbase + r) * CC + ocol] = f2bf(acc[r] + bs);
  }
}

// ---------------------------------------------------------------------------
// 5) BatchNorm (training) statistics + apply
// ---------------------------------------------------------------------------
__global__ void bn_stats(const unsigned short* __restrict__ v,
                         float* __restrict__ sum, float* __restrict__ sq) {
  const int c = threadIdx.x;                // 256 channels
  const size_t r0 = (size_t)blockIdx.x * 256;
  float s = 0.f, s2 = 0.f;
  for (int r = 0; r < 256; ++r) {
    float x = bf2f(v[(r0 + r) * CC + c]);
    s += x; s2 += x * x;
  }
  atomicAdd(&sum[c], s);
  atomicAdd(&sq[c], s2);
}

__global__ void bn_apply(unsigned short* __restrict__ v,
                         const float* __restrict__ sum, const float* __restrict__ sq,
                         const float* __restrict__ gamma, const float* __restrict__ beta) {
  const int c = threadIdx.x;
  const float invM = 1.0f / (float)MM;
  const float mean = sum[c] * invM;
  const float var = sq[c] * invM - mean * mean;
  const float sc = gamma[c] * rsqrtf(var + 1e-5f);
  const float sh = beta[c] - mean * sc;
  const size_t r0 = (size_t)blockIdx.x * 256;
  for (int r = 0; r < 256; ++r) {
    size_t idx = (r0 + r) * CC + c;
    v[idx] = f2bf(bf2f(v[idx]) * sc + sh);
  }
}

// ---------------------------------------------------------------------------
// 6) Banded flash attention. Block = 8 waves = one (n, 128-query strip).
//    K tile staged by async global->LDS DMA; V tile transposed via registers.
// ---------------------------------------------------------------------------
#define KT 32
#define KLDS_STRIDE (CC + 8)     // 264 u16 -> 528 B row stride (16B aligned)
#define VLDS_STRIDE (KT + 16)    // 48 u16  -> 96 B row stride (16B aligned)

__global__ __launch_bounds__(256, 1) void attention_kernel(
    const unsigned short* __restrict__ Qm,
    const unsigned short* __restrict__ Km,
    const unsigned short* __restrict__ Vm,
    unsigned short* __restrict__ Ym) {
  __shared__ unsigned short k_lds[KT * KLDS_STRIDE];
  __shared__ unsigned short v_lds[CC * VLDS_STRIDE];
  __shared__ unsigned short p_lds[8 * 16 * KT];

  const int n = blockIdx.y;                 // 0..127
  const int q0blk = blockIdx.x * 128;       // query strip base
  const int wave = threadIdx.x >> 5;
  const int lane = threadIdx.x & 31;
  const int q0 = q0blk + wave * 16;
  const int nk = lane & 15;
  const int hi = (lane < 16) ? 0 : 1;

  FragAB qa[8];
  {
    const uint4* rp = (const uint4*)(Qm + ((size_t)n * WW + q0 + nk) * CC);
    #pragma unroll
    for (int kc = 0; kc < 8; ++kc) {
      qa[kc].u4[0] = rp[kc * 4 + hi * 2];
      qa[kc].u4[1] = rp[kc * 4 + hi * 2 + 1];
    }
  }

  v8f acc[16];
  #pragma unroll
  for (int cg = 0; cg < 16; ++cg) acc[cg] = vzero();
  float mrow[8], lrow[8];
  #pragma unroll
  for (int r = 0; r < 8; ++r) { mrow[r] = -1.0e30f; lrow[r] = 0.0f; }

  int kstart = q0blk - (BAND - 1); if (kstart < 0) kstart = 0; kstart &= ~31;
  int kend = q0blk + 127 + BAND; if (kend > WW) kend = WW;
  kend = (kend + 31) & ~31; if (kend > WW) kend = WW;

  const int ldr = threadIdx.x >> 3;         // staging key row 0..31
  const int ldc = threadIdx.x & 7;          // staging 32-channel chunk
  // per-thread LDS destination offset for the async K-tile copy (64 B each)
  const unsigned kdst = (unsigned)(size_t)(&k_lds[ldr * KLDS_STRIDE + ldc * 32]);

  for (int jt = kstart; jt < kend; jt += KT) {
    {
      // K tile: async DMA straight into LDS, no VGPR round trip.
      async_copy_64B(kdst, Km + ((size_t)n * WW + jt + ldr) * CC + ldc * 32);

      // prefetch next tile's K/V lines while this tile is consumed
      if (jt + KT < kend) {
        __builtin_prefetch(Km + ((size_t)n * WW + jt + KT + ldr) * CC + ldc * 32, 0, 0);
        __builtin_prefetch(Vm + ((size_t)n * WW + jt + KT + ldr) * CC + ldc * 32, 0, 0);
      }

      // V tile: transposed (channel-major) staging via registers.
      const uint4* vs = (const uint4*)(Vm + ((size_t)n * WW + jt + ldr) * CC + ldc * 32);
      union { uint4 q4[4]; unsigned short s[32]; } vv;
      vv.q4[0] = vs[0]; vv.q4[1] = vs[1]; vv.q4[2] = vs[2]; vv.q4[3] = vs[3];
      #pragma unroll
      for (int i = 0; i < 32; ++i)
        v_lds[(ldc * 32 + i) * VLDS_STRIDE + ldr] = vv.s[i];

      wait_async0();               // K tile landed in LDS
    }
    __syncthreads();

    const bool active = (jt + KT > q0 - (BAND - 1)) && (jt < q0 + 15 + BAND);
    if (active) {
      v8f slo = vzero(), shi = vzero();
      #pragma unroll
      for (int kc = 0; kc < 8; ++kc) {
        FragAB blo, bhi;
        const uint4* kl = (const uint4*)(&k_lds[nk * KLDS_STRIDE + kc * 32 + hi * 16]);
        blo.u4[0] = kl[0]; blo.u4[1] = kl[1];
        const uint4* kh = (const uint4*)(&k_lds[(nk + 16) * KLDS_STRIDE + kc * 32 + hi * 16]);
        bhi.u4[0] = kh[0]; bhi.u4[1] = kh[1];
        slo = wmma_bf16(qa[kc], blo, slo);
        shi = wmma_bf16(qa[kc], bhi, shi);
      }

      const int qrow0 = q0 + hi * 8;
      float p_lo[8], p_hi[8], corr[8];
      #pragma unroll
      for (int r = 0; r < 8; ++r) {
        const int qr = qrow0 + r;
        int dlo = qr - (jt + nk);      dlo = dlo < 0 ? -dlo : dlo;
        int dhi = qr - (jt + 16 + nk); dhi = dhi < 0 ? -dhi : dhi;
        float vlo = (dlo < BAND) ? slo[r] : -1.0e30f;
        float vhi = (dhi < BAND) ? shi[r] : -1.0e30f;
        float mt = fmaxf(vlo, vhi);
        #pragma unroll
        for (int off = 8; off >= 1; off >>= 1)
          mt = fmaxf(mt, __shfl_xor(mt, off, 16));
        const float mnew = fmaxf(mrow[r], mt);
        corr[r] = __expf(mrow[r] - mnew);
        const float plo = __expf(vlo - mnew);
        const float phi = __expf(vhi - mnew);
        float ps = plo + phi;
        #pragma unroll
        for (int off = 8; off >= 1; off >>= 1)
          ps += __shfl_xor(ps, off, 16);
        lrow[r] = lrow[r] * corr[r] + ps;
        mrow[r] = mnew;
        p_lo[r] = plo; p_hi[r] = phi;
      }
      #pragma unroll
      for (int cg = 0; cg < 16; ++cg) {
        #pragma unroll
        for (int r = 0; r < 8; ++r) acc[cg][r] *= corr[r];
      }

      unsigned short* pw = &p_lds[wave * 16 * KT];
      #pragma unroll
      for (int r = 0; r < 8; ++r) {
        const int row = r + hi * 8;
        pw[row * KT + nk] = f2bf(p_lo[r]);
        pw[row * KT + 16 + nk] = f2bf(p_hi[r]);
      }
      FragAB pa;
      {
        const uint4* pr = (const uint4*)(pw + nk * KT);
        pa.u4[0] = pr[hi * 2];
        pa.u4[1] = pr[hi * 2 + 1];
      }

      #pragma unroll
      for (int cg = 0; cg < 16; ++cg) {
        FragAB vb;
        const uint4* vp = (const uint4*)(&v_lds[(cg * 16 + nk) * VLDS_STRIDE + hi * 16]);
        vb.u4[0] = vp[0]; vb.u4[1] = vp[1];
        acc[cg] = wmma_bf16(pa, vb, acc[cg]);
      }
    }
    __syncthreads();
  }

  float linv[8];
  #pragma unroll
  for (int r = 0; r < 8; ++r) linv[r] = 1.0f / lrow[r];
  const int rbase = q0 + hi * 8;
  #pragma unroll
  for (int cg = 0; cg < 16; ++cg) {
    const int col = cg * 16 + nk;
    #pragma unroll
    for (int r = 0; r < 8; ++r)
      Ym[((size_t)n * WW + rbase + r) * CC + col] = f2bf(acc[cg][r] * linv[r]);
  }
}

// ---------------------------------------------------------------------------
// 7) final: BN(zpre) + residual, transpose back to (B,C,H,W) f32
// ---------------------------------------------------------------------------
__global__ void final_out(const unsigned short* __restrict__ zp,
                          const float* __restrict__ sum, const float* __restrict__ sq,
                          const float* __restrict__ gamma, const float* __restrict__ beta,
                          const float* __restrict__ xres, float* __restrict__ out) {
  __shared__ float tile[32][33];
  const int bh = blockIdx.z;
  const int b = bh >> 6, h = bh & 63;
  const int c0 = blockIdx.y * 32;
  const int w0 = blockIdx.x * 32;
  const int tx = threadIdx.x, ty = threadIdx.y;
  #pragma unroll
  for (int i = 0; i < 4; ++i) {
    int wl = ty + i * 8;
    tile[wl][tx] = bf2f(zp[((size_t)bh * WW + w0 + wl) * CC + c0 + tx]);
  }
  __syncthreads();
  const float invM = 1.0f / (float)MM;
  #pragma unroll
  for (int i = 0; i < 4; ++i) {
    const int c = c0 + ty + i * 8;
    const float mean = sum[c] * invM;
    const float var = sq[c] * invM - mean * mean;
    const float sc = gamma[c] * rsqrtf(var + 1e-5f);
    const float sh = beta[c] - mean * sc;
    const size_t oidx = (((size_t)b * CC + c) * HH + h) * WW + w0 + tx;
    out[oidx] = tile[tx][ty + i * 8] * sc + sh + xres[oidx];
  }
}

// ---------------------------------------------------------------------------
// launch
// ---------------------------------------------------------------------------
extern "C" void kernel_launch(void* const* d_in, const int* in_sizes, int n_in,
                              void* d_out, int out_size, void* d_ws, size_t ws_size,
                              hipStream_t stream) {
  const float* x1    = (const float*)d_in[0];
  const float* x2    = (const float*)d_in[1];
  const float* Qw    = (const float*)d_in[2];
  const float* Qb    = (const float*)d_in[3];
  const float* Kw    = (const float*)d_in[4];
  const float* Kb    = (const float*)d_in[5];
  const float* Vw    = (const float*)d_in[6];
  const float* Vb    = (const float*)d_in[7];
  const float* Vg    = (const float*)d_in[8];
  const float* Vbeta = (const float*)d_in[9];
  const float* Ww    = (const float*)d_in[10];
  const float* Wb    = (const float*)d_in[11];
  const float* Wg    = (const float*)d_in[12];
  const float* Wbeta = (const float*)d_in[13];
  float* out = (float*)d_out;
  const size_t tensor_elems = (size_t)BB * CC * HH * WW;   // 16,777,216

  char* ws = (char*)d_ws;
  size_t off = 0;
  auto alloc = [&](size_t bytes) -> char* {
    char* p = ws + off;
    off += (bytes + 255) & ~(size_t)255;
    return p;
  };
  const size_t MC2 = (size_t)MM * CC * 2;
  unsigned short* xt1 = (unsigned short*)alloc(MC2);
  unsigned short* xt2 = (unsigned short*)alloc(MC2);
  unsigned short* wqb = (unsigned short*)alloc((size_t)CC * CC * 2);
  unsigned short* wkb = (unsigned short*)alloc((size_t)CC * CC * 2);
  unsigned short* wvb = (unsigned short*)alloc((size_t)CC * CC * 2);
  unsigned short* wwb = (unsigned short*)alloc((size_t)CC * CC * 2);
  unsigned short* q1  = (unsigned short*)alloc(MC2);
  unsigned short* k1  = (unsigned short*)alloc(MC2);
  unsigned short* v1  = (unsigned short*)alloc(MC2);
  unsigned short* q2  = (unsigned short*)alloc(MC2);
  unsigned short* k2  = (unsigned short*)alloc(MC2);
  unsigned short* v2  = (unsigned short*)alloc(MC2);
  unsigned short* y1  = (unsigned short*)alloc(MC2);
  unsigned short* y2  = (unsigned short*)alloc(MC2);
  unsigned short* zp1 = (unsigned short*)alloc(MC2);
  unsigned short* zp2 = (unsigned short*)alloc(MC2);
  float* stats = (float*)alloc(8 * CC * sizeof(float));

  hipMemsetAsync(stats, 0, 8 * CC * sizeof(float), stream);

  convert_weights<<<256, 256, 0, stream>>>(Qw, Kw, Vw, Ww, wqb, wkb, wvb, wwb);

  dim3 tb(32, 8);
  dim3 tg(WW / 32, CC / 32, NN);
  transpose_to_bf16<<<tg, tb, 0, stream>>>(x1, xt1);
  transpose_to_bf16<<<tg, tb, 0, stream>>>(x2, xt2);

  qkv_gemm<<<512, 256, 0, stream>>>(xt1, wqb, Qb, q1, wkb, Kb, k1, wvb, Vb, v1);
  qkv_gemm<<<512, 256, 0, stream>>>(xt2, wqb, Qb, q2, wkb, Kb, k2, wvb, Vb, v2);

  bn_stats<<<256, 256, 0, stream>>>(v1, stats + 0 * CC, stats + 1 * CC);
  bn_stats<<<256, 256, 0, stream>>>(v2, stats + 2 * CC, stats + 3 * CC);
  bn_apply<<<256, 256, 0, stream>>>(v1, stats + 0 * CC, stats + 1 * CC, Vg, Vbeta);
  bn_apply<<<256, 256, 0, stream>>>(v2, stats + 2 * CC, stats + 3 * CC, Vg, Vbeta);

  attention_kernel<<<dim3(4, NN), 256, 0, stream>>>(q1, k2, v2, y1);
  attention_kernel<<<dim3(4, NN), 256, 0, stream>>>(q2, k1, v1, y2);

  gemm_one<<<512, 256, 0, stream>>>(y1, wwb, Wb, zp1);
  gemm_one<<<512, 256, 0, stream>>>(y2, wwb, Wb, zp2);

  bn_stats<<<256, 256, 0, stream>>>(zp1, stats + 4 * CC, stats + 5 * CC);
  bn_stats<<<256, 256, 0, stream>>>(zp2, stats + 6 * CC, stats + 7 * CC);

  final_out<<<tg, tb, 0, stream>>>(zp1, stats + 4 * CC, stats + 5 * CC, Wg, Wbeta, x1, out);
  final_out<<<tg, tb, 0, stream>>>(zp2, stats + 6 * CC, stats + 7 * CC, Wg, Wbeta, x2, out + tensor_elems);
}
